// MultiHeadAttentionBlock_70411693850668
// MI455X (gfx1250) — compile-verified, matrix operand order checked by reading, not run on hardware
//
#include <hip/hip_runtime.h>
#include <hip/hip_bf16.h>

typedef __attribute__((ext_vector_type(16))) __bf16 v16bf;
typedef __attribute__((ext_vector_type(8)))  float  v8f;

#define D_MODEL 1024
#define NHEAD   16
#define DKH     64
#define SEQ     2048
#define BATCH   2
#define MTOT    (BATCH * SEQ)   // 4096

static __device__ __forceinline__ v8f vzero8() {
    v8f z = {0.f, 0.f, 0.f, 0.f, 0.f, 0.f, 0.f, 0.f};
    return z;
}

union Frag16 {                 // one 16x16x32 bf16 WMMA A/B operand
    v16bf  v;
    float4 q[2];
};

// LDS byte offset of a generic pointer into __shared__ (low 32 bits = LDS addr)
static __device__ __forceinline__ unsigned lds_off(const void* p) {
    return (unsigned)(uintptr_t)p;
}

// 32B per lane: two async 16B copies, global -> LDS (ASYNCcnt += 2)
static __device__ __forceinline__ void async_cp32(unsigned lds,
                                                  const __bf16* g) {
    asm volatile(
        "global_load_async_to_lds_b128 %0, %1, off\n\t"
        "global_load_async_to_lds_b128 %0, %1, off offset:16"
        :: "v"(lds), "v"((unsigned long long)(uintptr_t)g) : "memory");
}

// 64B per lane: four async 16B copies, global -> LDS (ASYNCcnt += 4)
static __device__ __forceinline__ void async_cp64(unsigned lds,
                                                  const __bf16* g) {
    asm volatile(
        "global_load_async_to_lds_b128 %0, %1, off\n\t"
        "global_load_async_to_lds_b128 %0, %1, off offset:16\n\t"
        "global_load_async_to_lds_b128 %0, %1, off offset:32\n\t"
        "global_load_async_to_lds_b128 %0, %1, off offset:48"
        :: "v"(lds), "v"((unsigned long long)(uintptr_t)g) : "memory");
}

// ---------------------------------------------------------------------------
// fp32 -> bf16 conversion (vectorized by 4; all sizes are multiples of 1024)
// ---------------------------------------------------------------------------
__global__ void cvt_f32_to_bf16(const float* __restrict__ in,
                                __bf16* __restrict__ out, int n4) {
    int i = blockIdx.x * blockDim.x + threadIdx.x;
    if (i < n4) {
        float4 f = reinterpret_cast<const float4*>(in)[i];
        __bf16* o = out + (size_t)i * 4;
        o[0] = (__bf16)f.x; o[1] = (__bf16)f.y;
        o[2] = (__bf16)f.z; o[3] = (__bf16)f.w;
    }
}

// ---------------------------------------------------------------------------
// GEMM: C[M,N] = A_bf16[M,K] @ W_bf16[N,K]^T + bias  (K = N = 1024)
// mode 0: write bf16 scattered to [B, H, S, DKH]   (QKV projections)
// mode 1: write fp32 row-major [M, N]              (output projection)
// Block tile 128x128, 8 waves (2 M x 4 N), wave tile 64x32.
// Double-buffered LDS fed by global_load_async_to_lds_b128 (ASYNCcnt).
// ---------------------------------------------------------------------------
__global__ __launch_bounds__(256)
void gemm_bf16_nt(const __bf16* __restrict__ A,
                  const __bf16* __restrict__ W,
                  const float* __restrict__ bias,
                  void* __restrict__ outp, int mode) {
    constexpr int K = 1024;
    constexpr int LDT = 40;                 // LDS row stride (bf16), 16B aligned
    __shared__ __bf16 ldsA[2][128 * LDT];
    __shared__ __bf16 ldsB[2][128 * LDT];

    const int tid  = threadIdx.x;
    const int lane = tid & 31;
    const int wave = tid >> 5;
    const int wy   = wave >> 2;             // 0..1 -> M offset
    const int wx   = wave & 3;              // 0..3 -> N offset
    const int half = lane >> 4;             // K-half selector (CDNA5 layout)
    const int l16  = lane & 15;

    const int mBlock = blockIdx.y * 128;
    const int nBlock = blockIdx.x * 128;

    // cooperative staging: 2 threads per 32-wide row, 16 bf16 (32B) each
    const int rowL = tid >> 1;
    const int colH = (tid & 1) * 16;
    const __bf16* agBase = A + (size_t)(mBlock + rowL) * K + colH;
    const __bf16* wgBase = W + (size_t)(nBlock + rowL) * K + colH;

    v8f acc[4][2];
    for (int mi = 0; mi < 4; ++mi)
        for (int ni = 0; ni < 2; ++ni) acc[mi][ni] = vzero8();

    // prologue: async-stage first K tile into buffer 0
    async_cp32(lds_off(&ldsA[0][rowL * LDT + colH]), agBase);
    async_cp32(lds_off(&ldsB[0][rowL * LDT + colH]), wgBase);

    int buf = 0;
    for (int k0 = 0; k0 < K; k0 += 32) {
        if (k0 + 32 < K) {
            // prefetch next tile into the other buffer, then wait for current
            async_cp32(lds_off(&ldsA[buf ^ 1][rowL * LDT + colH]),
                       agBase + k0 + 32);
            async_cp32(lds_off(&ldsB[buf ^ 1][rowL * LDT + colH]),
                       wgBase + k0 + 32);
            asm volatile("s_wait_asynccnt 0x4" ::: "memory");
        } else {
            asm volatile("s_wait_asynccnt 0x0" ::: "memory");
        }
        __syncthreads();

        // A fragments: lane<16 holds K 0-7 & 16-23; lane>=16 holds K 8-15 & 24-31
        Frag16 af[4], bfr[2];
        for (int mi = 0; mi < 4; ++mi) {
            const __bf16* p =
                &ldsA[buf][(wy * 64 + mi * 16 + l16) * LDT + half * 8];
            af[mi].q[0] = *reinterpret_cast<const float4*>(p);
            af[mi].q[1] = *reinterpret_cast<const float4*>(p + 16);
        }
        for (int ni = 0; ni < 2; ++ni) {
            const __bf16* p =
                &ldsB[buf][(wx * 32 + ni * 16 + l16) * LDT + half * 8];
            bfr[ni].q[0] = *reinterpret_cast<const float4*>(p);
            bfr[ni].q[1] = *reinterpret_cast<const float4*>(p + 16);
        }
        for (int mi = 0; mi < 4; ++mi)
            for (int ni = 0; ni < 2; ++ni)
                acc[mi][ni] = __builtin_amdgcn_wmma_f32_16x16x32_bf16(
                    false, af[mi].v, false, bfr[ni].v,
                    (short)0, acc[mi][ni], false, false);

        __syncthreads();                    // buffer consumed by all waves
        buf ^= 1;
    }

    // epilogue: C/D layout -> VGPR r holds row r (lanes 0-15) / r+8 (lanes 16-31)
    for (int mi = 0; mi < 4; ++mi)
        for (int ni = 0; ni < 2; ++ni)
            for (int r = 0; r < 8; ++r) {
                int row = mBlock + wy * 64 + mi * 16 + r + half * 8;
                int col = nBlock + wx * 32 + ni * 16 + l16;
                float v = acc[mi][ni][r] + bias[col];
                if (mode == 0) {
                    int b = row >> 11, s = row & 2047;
                    int h = col >> 6,  d = col & 63;
                    ((__bf16*)outp)[(((size_t)b * NHEAD + h) * SEQ + s) * DKH + d] =
                        (__bf16)v;
                } else {
                    ((float*)outp)[(size_t)row * D_MODEL + col] = v;
                }
            }
}

// ---------------------------------------------------------------------------
// Flash attention: one block = 64 query rows of one (b,h); 4 waves x 16 rows.
// K tile async-staged to LDS; V transposed through registers; online softmax.
// Writes attention output bf16 directly in concat layout [B, S, H*DKH].
// ---------------------------------------------------------------------------
__global__ __launch_bounds__(128)
void flash_attn_bf16(const __bf16* __restrict__ Qp,
                     const __bf16* __restrict__ Kp,
                     const __bf16* __restrict__ Vp,
                     const int* __restrict__ mask,
                     __bf16* __restrict__ A2) {
    constexpr int LDT = 72;                     // LDS stride (bf16), 16B aligned
    __shared__ __bf16 ldsK[64 * LDT];           // [key][dk]
    __shared__ __bf16 ldsV[64 * LDT];           // transposed: [dk][key]
    __shared__ __bf16 ldsP[4][16 * LDT];        // per-wave P scratch [row][key]

    const int bh   = blockIdx.y;                // 0..B*H-1
    const int b    = bh >> 4;
    const int h    = bh & 15;
    const int q0   = blockIdx.x * 64;
    const int tid  = threadIdx.x;
    const int lane = tid & 31;
    const int wave = tid >> 5;
    const int half = lane >> 4;
    const int l16  = lane & 15;

    const __bf16* Qh = Qp + (size_t)bh * SEQ * DKH;
    const __bf16* Kh = Kp + (size_t)bh * SEQ * DKH;
    const __bf16* Vh = Vp + (size_t)bh * SEQ * DKH;
    const int* maskb = mask + (size_t)b * SEQ * SEQ;

    // Q fragments for this wave's 16 rows (A-layout, 2 k-chunks over DKH=64)
    Frag16 qf[2];
    {
        const __bf16* p = Qh + (size_t)(q0 + wave * 16 + l16) * DKH;
        for (int kc = 0; kc < 2; ++kc) {
            qf[kc].q[0] = *reinterpret_cast<const float4*>(p + kc * 32 + half * 8);
            qf[kc].q[1] = *reinterpret_cast<const float4*>(p + kc * 32 + half * 8 + 16);
        }
    }

    v8f o[4];
    for (int nb = 0; nb < 4; ++nb) o[nb] = vzero8();
    float mrun[8], lrun[8];
    for (int r = 0; r < 8; ++r) { mrun[r] = -1e30f; lrun[r] = 0.f; }

    const float scale = 0.125f;                 // 1/sqrt(64)
    const int rowS = tid >> 1;                  // staging: key row 0..63
    const int chS  = (tid & 1) * 32;            // staging: dk base 0 or 32

    for (int kt = 0; kt < SEQ; kt += 64) {
        // ---- stage tiles: K via async-to-LDS, V transposed via registers ----
        {
            union { float4 f[4]; __bf16 e[32]; } vv;
            const __bf16* vg = Vh + (size_t)(kt + rowS) * DKH + chS;
            vv.f[0] = *reinterpret_cast<const float4*>(vg);
            vv.f[1] = *reinterpret_cast<const float4*>(vg + 8);
            vv.f[2] = *reinterpret_cast<const float4*>(vg + 16);
            vv.f[3] = *reinterpret_cast<const float4*>(vg + 24);

            __syncthreads();                    // previous tile fully consumed
            async_cp64(lds_off(&ldsK[rowS * LDT + chS]),
                       Kh + (size_t)(kt + rowS) * DKH + chS);
            for (int j = 0; j < 32; ++j)        // transpose V into [dk][key]
                ldsV[(chS + j) * LDT + rowS] = vv.e[j];
            asm volatile("s_wait_asynccnt 0x0" ::: "memory");
            __syncthreads();
        }
        if (kt + 64 < SEQ)
            __builtin_prefetch(Vh + (size_t)(kt + 64) * DKH, 0, 3);

        // ---- scores S = Q @ K^T for 4 key sub-tiles of 16 ----
        v8f s[4];
        for (int kb = 0; kb < 4; ++kb) {
            Frag16 kf[2];
            const __bf16* base = &ldsK[(kb * 16 + l16) * LDT + half * 8];
            for (int kc = 0; kc < 2; ++kc) {
                kf[kc].q[0] = *reinterpret_cast<const float4*>(base + kc * 32);
                kf[kc].q[1] = *reinterpret_cast<const float4*>(base + kc * 32 + 16);
            }
            v8f z = vzero8();
            z = __builtin_amdgcn_wmma_f32_16x16x32_bf16(false, qf[0].v, false,
                                                        kf[0].v, (short)0, z,
                                                        false, false);
            z = __builtin_amdgcn_wmma_f32_16x16x32_bf16(false, qf[1].v, false,
                                                        kf[1].v, (short)0, z,
                                                        false, false);
            s[kb] = z;
        }

        // ---- scale + mask + online softmax ----
        float mtile[8];
        for (int r = 0; r < 8; ++r) mtile[r] = -1e30f;
        for (int kb = 0; kb < 4; ++kb)
            for (int r = 0; r < 8; ++r) {
                float val = s[kb][r] * scale;
                int qrow = q0 + wave * 16 + r + half * 8;
                int kcol = kt + kb * 16 + l16;
                if (maskb[(size_t)qrow * SEQ + kcol] == 0) val = -1e9f;
                s[kb][r] = val;
                mtile[r] = fmaxf(mtile[r], val);
            }
        for (int r = 0; r < 8; ++r)
            for (int off = 1; off < 16; off <<= 1)
                mtile[r] = fmaxf(mtile[r], __shfl_xor(mtile[r], off, 16));

        float alpha[8], rsum[8];
        for (int r = 0; r < 8; ++r) {
            float mnew = fmaxf(mrun[r], mtile[r]);
            alpha[r] = __expf(mrun[r] - mnew);
            mrun[r]  = mnew;
            rsum[r]  = 0.f;
        }
        for (int kb = 0; kb < 4; ++kb)
            for (int r = 0; r < 8; ++r) {
                float p = __expf(s[kb][r] - mrun[r]);
                s[kb][r] = p;
                rsum[r] += p;
            }
        for (int r = 0; r < 8; ++r) {
            for (int off = 1; off < 16; off <<= 1)
                rsum[r] += __shfl_xor(rsum[r], off, 16);
            lrun[r] = lrun[r] * alpha[r] + rsum[r];
        }
        for (int nb = 0; nb < 4; ++nb)
            for (int r = 0; r < 8; ++r) o[nb][r] *= alpha[r];

        // ---- relayout P (C-layout -> A-layout) through per-wave LDS ----
        __bf16* Pw = ldsP[wave];
        for (int kb = 0; kb < 4; ++kb)
            for (int r = 0; r < 8; ++r)
                Pw[(r + half * 8) * LDT + kb * 16 + l16] = (__bf16)s[kb][r];
        __syncthreads();                        // order DS write -> DS read

        Frag16 pf[2];
        for (int kc = 0; kc < 2; ++kc) {
            const __bf16* p = &Pw[l16 * LDT + kc * 32 + half * 8];
            pf[kc].q[0] = *reinterpret_cast<const float4*>(p);
            pf[kc].q[1] = *reinterpret_cast<const float4*>(p + 16);
        }

        // ---- O += P @ V ----
        for (int nb = 0; nb < 4; ++nb) {
            Frag16 vf[2];
            const __bf16* base = &ldsV[(nb * 16 + l16) * LDT + half * 8];
            for (int kc = 0; kc < 2; ++kc) {
                vf[kc].q[0] = *reinterpret_cast<const float4*>(base + kc * 32);
                vf[kc].q[1] = *reinterpret_cast<const float4*>(base + kc * 32 + 16);
            }
            o[nb] = __builtin_amdgcn_wmma_f32_16x16x32_bf16(false, pf[0].v, false,
                                                            vf[0].v, (short)0,
                                                            o[nb], false, false);
            o[nb] = __builtin_amdgcn_wmma_f32_16x16x32_bf16(false, pf[1].v, false,
                                                            vf[1].v, (short)0,
                                                            o[nb], false, false);
        }
        __syncthreads();                        // protect ldsP/K/V before next iter
    }

    // ---- normalize and store in concat layout [B, S, H*DKH] (bf16) ----
    for (int nb = 0; nb < 4; ++nb)
        for (int r = 0; r < 8; ++r) {
            int qrow = q0 + wave * 16 + r + half * 8;
            int dcol = h * DKH + nb * 16 + l16;
            float val = o[nb][r] / lrun[r];
            A2[((size_t)b * SEQ + qrow) * D_MODEL + dcol] = (__bf16)val;
        }
}

// ---------------------------------------------------------------------------
// Host launch
// ---------------------------------------------------------------------------
extern "C" void kernel_launch(void* const* d_in, const int* in_sizes, int n_in,
                              void* d_out, int out_size, void* d_ws, size_t ws_size,
                              hipStream_t stream) {
    const float* q    = (const float*)d_in[0];
    const float* k    = (const float*)d_in[1];
    const float* v    = (const float*)d_in[2];
    const int*   mask = (const int*)  d_in[3];
    const float* w_q  = (const float*)d_in[4];
    const float* b_q  = (const float*)d_in[5];
    const float* w_k  = (const float*)d_in[6];
    const float* b_k  = (const float*)d_in[7];
    const float* w_v  = (const float*)d_in[8];
    const float* b_v  = (const float*)d_in[9];
    const float* w_o  = (const float*)d_in[10];
    const float* b_o  = (const float*)d_in[11];

    const size_t NX = (size_t)MTOT * D_MODEL;      // 4096*1024
    const size_t NW = (size_t)D_MODEL * D_MODEL;   // 1024*1024

    __bf16* ws  = (__bf16*)d_ws;
    size_t  off = 0;
    auto take = [&](size_t n) { __bf16* p = ws + off; off += n; return p; };
    __bf16* qb  = take(NX);
    __bf16* kb  = take(NX);
    __bf16* vb  = take(NX);
    __bf16* wqb = take(NW);
    __bf16* wkb = take(NW);
    __bf16* wvb = take(NW);
    __bf16* wob = take(NW);
    __bf16* Qp  = take(NX);   // [B,H,S,DKH]
    __bf16* Kp  = take(NX);
    __bf16* Vp  = take(NX);
    __bf16* A2  = take(NX);   // concat attention output [B,S,D_MODEL]

    auto cvt = [&](const float* src, __bf16* dst, size_t n) {
        int n4 = (int)(n / 4);
        cvt_f32_to_bf16<<<(n4 + 255) / 256, 256, 0, stream>>>(src, dst, n4);
    };
    cvt(q, qb, NX);   cvt(k, kb, NX);   cvt(v, vb, NX);
    cvt(w_q, wqb, NW); cvt(w_k, wkb, NW); cvt(w_v, wvb, NW); cvt(w_o, wob, NW);

    dim3 ggrid(D_MODEL / 128, MTOT / 128);         // (8, 32)
    gemm_bf16_nt<<<ggrid, 256, 0, stream>>>(qb, wqb, b_q, Qp, 0);
    gemm_bf16_nt<<<ggrid, 256, 0, stream>>>(kb, wkb, b_k, Kp, 0);
    gemm_bf16_nt<<<ggrid, 256, 0, stream>>>(vb, wvb, b_v, Vp, 0);

    dim3 agrid(SEQ / 64, BATCH * NHEAD);           // (32, 32)
    flash_attn_bf16<<<agrid, 128, 0, stream>>>(Qp, Kp, Vp, mask, A2);

    gemm_bf16_nt<<<ggrid, 256, 0, stream>>>(A2, wob, b_o, d_out, 1);
}
